// ConformerBlock_83588653515424
// MI455X (gfx1250) — matrix operand order repaired and checked by reading, not compiled, not run on hardware
//
#include <hip/hip_runtime.h>
#include <math.h>

typedef _Float16 h16;
typedef __attribute__((ext_vector_type(16))) _Float16 v16h;
typedef __attribute__((ext_vector_type(8)))  float    v8f;

#define B_   32
#define S_   512
#define D_   512
#define H_   8
#define DH_  64
#define MTOK (B_ * S_)          /* 16384 tokens */
#define KCONV 31

// ---- CDNA5 async global->LDS path (guarded; falls back to manual staging) ----
#if defined(__has_builtin)
#  if __has_builtin(__builtin_amdgcn_global_load_async_to_lds_b128)
#    define HAVE_ASYNC_LDS 1
#  endif
#  if __has_builtin(__builtin_amdgcn_s_wait_asynccnt)
#    define HAVE_WAIT_ASYNC 1
#  endif
#endif
#ifndef HAVE_ASYNC_LDS
#  define HAVE_ASYNC_LDS 0
#endif
#ifndef HAVE_WAIT_ASYNC
#  define HAVE_WAIT_ASYNC 0
#endif

#if HAVE_ASYNC_LDS
typedef int v4i_ __attribute__((vector_size(16)));
typedef __attribute__((address_space(1))) v4i_* g128p;
typedef __attribute__((address_space(3))) v4i_* l128p;
typedef __attribute__((address_space(1))) void* g0p;
typedef __attribute__((address_space(3))) void* l0p;
__device__ __forceinline__ void async_copy16(const void* g, void* l) {
    __builtin_amdgcn_global_load_async_to_lds_b128(
        (g128p)(g0p)const_cast<void*>(g),
        (l128p)(l0p)l, 0, 0);
}
__device__ __forceinline__ void wait_async0() {
#if HAVE_WAIT_ASYNC
    __builtin_amdgcn_s_wait_asynccnt(0);
#else
    asm volatile("s_wait_asynccnt 0" ::: "memory");
#endif
}
#endif

// ---------------------------------------------------------------------------
// Generic tiled WMMA GEMM:  C[z] = epilogue( A[z] (MxK) * B[z]^T )
// B is stored row-major transposed: element (k,n) at Bt + n*Kd + k  (f16)
// Block tile 128x64, 256 threads (8 waves); wave w owns rows [w*16, w*16+16)
// and all 64 cols -> 4 WMMA accumulators, A fragment reused 4x per K-step.
// A element (m,k)  at A  + zb*aZb + zh*aZh + m*lda + k            (TA = f32/f16)
// B element (k,n)  at Bt + zb*bZb + zh*bZh + n*Kd + k             (f16)
// C element (m,n)  at C  + zb*cZb + zh*cZh + m*ldc + n            (TO = f32/f16)
// epilogue: v = gemm (+bias[n]); if act==1 v=silu(v); if resid v=resid+scale*v
// ---------------------------------------------------------------------------
template <typename TA, typename TO>
__global__ __launch_bounds__(256) void gemm_wmma(
    const TA* __restrict__ A, const h16* __restrict__ Bt,
    const float* __restrict__ bias, const float* __restrict__ resid,
    TO* __restrict__ C,
    int M, int N, int Kd,
    int lda, long long aZb, long long aZh,
    long long bZb, long long bZh,
    int ldc, long long cZb, long long cZh,
    int act, float resScale, int Hn)
{
    const int z  = blockIdx.z;
    const int zb = z / Hn, zh = z % Hn;
    const TA*  Ab = A  + (long long)zb * aZb + (long long)zh * aZh;
    const h16* Bb = Bt + (long long)zb * bZb + (long long)zh * bZh;
    TO*        Cb = C  + (long long)zb * cZb + (long long)zh * cZh;
    const float* Rb = resid ? (resid + (long long)zb * cZb + (long long)zh * cZh)
                            : (const float*)nullptr;

    const int m0 = blockIdx.y * 128;
    const int n0 = blockIdx.x * 64;

    __shared__ h16 ldsA[128][48];   // 96B row stride: 16B-aligned async chunks
    __shared__ h16 ldsB[64][48];    // ldsB[n][k], same alignment property

    const int tid  = threadIdx.x;
    const int lane = tid & 31;
    const int w    = tid >> 5;      // wave 0..7 -> 16-row slice

    v8f acc[4] = {v8f{}, v8f{}, v8f{}, v8f{}};

    for (int k0 = 0; k0 < Kd; k0 += 32) {
        __syncthreads();                  // previous iteration fully consumed
        // ---- stage A tile (128 x 32) as f16 ----
        if constexpr (sizeof(TA) == 2) {
#if HAVE_ASYNC_LDS
            #pragma unroll
            for (int c = 0; c < 2; ++c) {         // 2 x 16B chunks per thread
                const int chunk = tid * 2 + c;    // 0..511
                const int row   = chunk >> 2;     // 0..127
                const int cg    = (chunk & 3) * 8;
                const TA* src = Ab + (long long)(m0 + row) * lda + k0 + cg;
                async_copy16((const void*)src, (void*)&ldsA[row][cg]);
            }
#else
            {
                const int row = tid >> 1, cb = (tid & 1) * 16;
                const TA* src = Ab + (long long)(m0 + row) * lda + k0 + cb;
                #pragma unroll
                for (int j = 0; j < 16; ++j) ldsA[row][cb + j] = src[j];
            }
#endif
        } else {
            const int row = tid >> 1, cb = (tid & 1) * 16;
            const TA* src = Ab + (long long)(m0 + row) * lda + k0 + cb;
            #pragma unroll
            for (int j = 0; j < 16; ++j) ldsA[row][cb + j] = (h16)(float)src[j];
        }
        // ---- stage B tile (64 n-rows x 32 k), contiguous in k ----
        {
            const int row = tid >> 2;             // 0..63
            const int cg  = (tid & 3) * 8;        // 0,8,16,24
            const h16* src = Bb + (long long)(n0 + row) * Kd + k0 + cg;
#if HAVE_ASYNC_LDS
            async_copy16((const void*)src, (void*)&ldsB[row][cg]);
#else
            #pragma unroll
            for (int j = 0; j < 8; ++j) ldsB[row][cg + j] = src[j];
#endif
        }
#if HAVE_ASYNC_LDS
        wait_async0();
#endif
        __syncthreads();

        // ---- fragments per ISA wave32 VGPR layouts ----
        v16h af;
        const int ml = w * 16 + (lane & 15);
        const int kb = (lane >> 4) * 8;
        #pragma unroll
        for (int j = 0; j < 8; ++j) {
            af[j]     = ldsA[ml][kb + j];
            af[8 + j] = ldsA[ml][kb + 16 + j];
        }
        const int kbb = (lane >> 4) * 8;
        #pragma unroll
        for (int ng = 0; ng < 4; ++ng) {
            v16h bf;
            const int nl = ng * 16 + (lane & 15);
            #pragma unroll
            for (int i = 0; i < 16; ++i) bf[i] = ldsB[nl][kbb + (i & 7) + ((i >> 3) << 4)];
            acc[ng] = __builtin_amdgcn_wmma_f32_16x16x32_f16(false, af, false, bf,
                                                             (short)0, acc[ng], false, false);
        }
    }

    // ---- epilogue: lane L -> N = L%16, VGPR r -> M = r + 8*(L/16) ----
    const int rbase = w * 16 + (lane >> 4) * 8;
    #pragma unroll
    for (int ng = 0; ng < 4; ++ng) {
        const int n = n0 + ng * 16 + (lane & 15);
        #pragma unroll
        for (int r = 0; r < 8; ++r) {
            const int m = m0 + rbase + r;
            float v0 = acc[ng][r];
            if (bias) v0 += bias[n];
            if (act == 1) v0 = v0 / (1.f + expf(-v0));
            const long long ci = (long long)m * ldc + n;
            if (Rb) v0 = Rb[ci] + resScale * v0;
            Cb[ci] = (TO)v0;
        }
    }
}

// ---------------------------------------------------------------------------
// Block reductions + LayerNorm (one 512-wide token per 256-thread block)
// ---------------------------------------------------------------------------
__device__ inline float block_sum(float v, float* red) {
    const int tid = threadIdx.x;
    red[tid] = v; __syncthreads();
    for (int s = 128; s > 0; s >>= 1) {
        if (tid < s) red[tid] += red[tid + s];
        __syncthreads();
    }
    float r = red[0]; __syncthreads();
    return r;
}

template <typename TO>
__global__ __launch_bounds__(256) void ln_kernel(const float* __restrict__ x,
                                                 const float* __restrict__ g,
                                                 const float* __restrict__ bb,
                                                 TO* __restrict__ y)
{
    __shared__ float red[256];
    const long long row = blockIdx.x;
    const float* xr = x + row * D_;
    const int tid = threadIdx.x;
    const float a0 = xr[tid], a1 = xr[tid + 256];
    const float mean = block_sum(a0 + a1, red) * (1.f / (float)D_);
    const float d0 = a0 - mean, d1 = a1 - mean;
    const float var = block_sum(d0 * d0 + d1 * d1, red) * (1.f / (float)D_);
    const float rstd = rsqrtf(var + 1e-5f);
    TO* yr = y + row * D_;
    yr[tid]       = (TO)(d0 * rstd * g[tid] + bb[tid]);
    yr[tid + 256] = (TO)(d1 * rstd * g[tid + 256] + bb[tid + 256]);
}

// ---------------------------------------------------------------------------
// Softmax over keys with fused Transformer-XL relative shift (gather from ps)
// ---------------------------------------------------------------------------
__global__ __launch_bounds__(256) void softmax_relshift(
    float* __restrict__ attn, const h16* __restrict__ ps, float scale)
{
    __shared__ float red[256];
    const long long gr = blockIdx.x;           // bh*S + i
    const long long bh = gr / S_;
    const int i = (int)(gr % S_);
    float* arow = attn + gr * S_;
    const h16* psb = ps + bh * (long long)(S_ * S_);
    const int tid = threadIdx.x;

    float v[2];
    #pragma unroll
    for (int e = 0; e < 2; ++e) {
        const int j  = tid + e * 256;
        const int f  = S_ * (i + 1) + j;       // pad/reshape/slice flat index
        const int q  = f / (S_ + 1);
        const int kp = f % (S_ + 1);
        const float pv = (kp == 0) ? 0.f : (float)psb[(long long)q * S_ + (kp - 1)];
        v[e] = (arow[j] + pv) * scale;
    }
    float m = fmaxf(v[0], v[1]);
    red[tid] = m; __syncthreads();
    for (int s = 128; s > 0; s >>= 1) {
        if (tid < s) red[tid] = fmaxf(red[tid], red[tid + s]);
        __syncthreads();
    }
    const float rowmax = red[0]; __syncthreads();
    const float e0 = expf(v[0] - rowmax), e1 = expf(v[1] - rowmax);
    const float inv = 1.f / block_sum(e0 + e1, red);
    arow[tid]       = e0 * inv;
    arow[tid + 256] = e1 * inv;
}

// ---------------------------------------------------------------------------
// Small elementwise helpers
// ---------------------------------------------------------------------------
// convert + transpose: src [K][N] f32 -> dst [N][K] f16
__global__ void f32_to_f16_T(const float* __restrict__ s, h16* __restrict__ d,
                             int K, int N) {
    const int i = blockIdx.x * 256 + threadIdx.x;
    if (i >= K * N) return;
    const int n = i / K, k = i % K;
    d[i] = (h16)s[(long long)k * N + n];
}

__global__ void pe_kernel(h16* __restrict__ pe, int n) {
    const int idx = blockIdx.x * 256 + threadIdx.x;
    if (idx >= n) return;
    const int c = idx % D_;
    const int s = idx / D_;
    const int ip = c >> 1;
    const float ang = (float)s * expf(-(float)(2 * ip) * (9.210340371976184f / (float)D_));
    pe[idx] = (h16)((c & 1) ? cosf(ang) : sinf(ang));
}

// relayout q/k/v [B,S,H*DH] -> qu/qv/kt: [B,H,S,DH], vT: [B,H,DH,S] (f16)
__global__ void qkv_prep(const float* __restrict__ q, const float* __restrict__ k,
                         const float* __restrict__ v, const float* __restrict__ u,
                         const float* __restrict__ vb,
                         h16* __restrict__ qu, h16* __restrict__ qv,
                         h16* __restrict__ kt, h16* __restrict__ vT, int n)
{
    const int idx = blockIdx.x * 256 + threadIdx.x;
    if (idx >= n) return;
    const int dh = idx % DH_;
    const int s  = (idx / DH_) % S_;
    const int h  = (idx / (DH_ * S_)) % H_;
    const int b  = idx / (DH_ * S_ * H_);
    const long long src = ((long long)(b * S_ + s)) * D_ + h * DH_ + dh;
    const float qq = q[src];
    qu[idx] = (h16)(qq + u[h * DH_ + dh]);
    qv[idx] = (h16)(qq + vb[h * DH_ + dh]);
    kt[idx] = (h16)k[src];
    vT[(((long long)b * H_ + h) * DH_ + dh) * S_ + s] = (h16)v[src];
}

__global__ void p_prep(const float* __restrict__ pf, h16* __restrict__ p16, int n) {
    const int idx = blockIdx.x * 256 + threadIdx.x;
    if (idx >= n) return;
    const int dh = idx % DH_;
    const int s  = (idx / DH_) % S_;
    const int h  = idx / (DH_ * S_);
    p16[idx] = (h16)pf[(long long)s * D_ + h * DH_ + dh];
}

__global__ void glu_kernel(const h16* __restrict__ big, float* __restrict__ out, int n) {
    const int idx = blockIdx.x * 256 + threadIdx.x;
    if (idx >= n) return;
    const int c = idx % D_;
    const long long m = idx / D_;
    const float a = (float)big[m * (2 * D_) + c];
    const float g = (float)big[m * (2 * D_) + D_ + c];
    out[idx] = a / (1.f + expf(-g));
}

__global__ void dwconv_bn_swish(const float* __restrict__ glu, const float* __restrict__ dw,
                                const float* __restrict__ bng, const float* __restrict__ bnb,
                                const float* __restrict__ bnm, const float* __restrict__ bnv,
                                h16* __restrict__ out, int n)
{
    const int idx = blockIdx.x * 256 + threadIdx.x;
    if (idx >= n) return;
    const int c = idx % D_;
    const int s = (idx / D_) % S_;
    const long long bbase = (long long)(idx / (D_ * S_)) * (long long)(D_ * S_);
    float acc = 0.f;
    #pragma unroll
    for (int t = 0; t < KCONV; ++t) {
        const int ss = s + t - (KCONV - 1) / 2;
        if (ss >= 0 && ss < S_)
            acc += glu[bbase + (long long)ss * D_ + c] * dw[c * KCONV + t];
    }
    acc = (acc - bnm[c]) * rsqrtf(bnv[c] + 1e-5f) * bng[c] + bnb[c];
    acc = acc / (1.f + expf(-acc));          // swish
    out[idx] = (h16)acc;
}

// ---------------------------------------------------------------------------
// Launcher
// ---------------------------------------------------------------------------
extern "C" void kernel_launch(void* const* d_in, const int* in_sizes, int n_in,
                              void* d_out, int out_size, void* d_ws, size_t ws_size,
                              hipStream_t stream)
{
    (void)in_sizes; (void)n_in; (void)out_size; (void)ws_size;
    const float* x      = (const float*)d_in[0];
    const float* ff1_g  = (const float*)d_in[1];
    const float* ff1_b  = (const float*)d_in[2];
    const float* ff1_w1 = (const float*)d_in[3];
    const float* ff1_b1 = (const float*)d_in[4];
    const float* ff1_w2 = (const float*)d_in[5];
    const float* ff1_b2 = (const float*)d_in[6];
    const float* attn_g = (const float*)d_in[7];
    const float* attn_b = (const float*)d_in[8];
    const float* wq     = (const float*)d_in[9];
    const float* bq     = (const float*)d_in[10];
    const float* wk     = (const float*)d_in[11];
    const float* bk     = (const float*)d_in[12];
    const float* wv     = (const float*)d_in[13];
    const float* bv     = (const float*)d_in[14];
    const float* wpos   = (const float*)d_in[15];
    const float* u_bias = (const float*)d_in[16];
    const float* v_bias = (const float*)d_in[17];
    const float* wo     = (const float*)d_in[18];
    const float* bo     = (const float*)d_in[19];
    const float* conv_g = (const float*)d_in[20];
    const float* conv_b = (const float*)d_in[21];
    const float* pw1_w  = (const float*)d_in[22];
    const float* pw1_b  = (const float*)d_in[23];
    const float* dw_w   = (const float*)d_in[24];
    const float* bn_g   = (const float*)d_in[25];
    const float* bn_b   = (const float*)d_in[26];
    const float* bn_m   = (const float*)d_in[27];
    const float* bn_v   = (const float*)d_in[28];
    const float* pw2_w  = (const float*)d_in[29];
    const float* pw2_b  = (const float*)d_in[30];
    const float* ff2_g  = (const float*)d_in[31];
    const float* ff2_b  = (const float*)d_in[32];
    const float* ff2_w1 = (const float*)d_in[33];
    const float* ff2_b1 = (const float*)d_in[34];
    const float* ff2_w2 = (const float*)d_in[35];
    const float* ff2_b2 = (const float*)d_in[36];
    const float* ln_g   = (const float*)d_in[37];
    const float* ln_b   = (const float*)d_in[38];

    float* outPtr  = (float*)d_out;                            // [B,S,D]
    float* attnOut = outPtr + (long long)B_ * S_ * D_;         // [B,H,S,S]

    // ---- bump allocator over workspace ----
    char* wsp = (char*)d_ws;
    auto alloc = [&](size_t bytes) -> void* {
        void* r = (void*)wsp;
        wsp += (bytes + 255) & ~(size_t)255;
        return r;
    };
    const size_t MD  = (size_t)MTOK * D_;       // 8.39M
    h16* w16_ff1w1 = (h16*)alloc(sizeof(h16) * D_ * 4 * D_);   // all stored [N][K]
    h16* w16_ff1w2 = (h16*)alloc(sizeof(h16) * 4 * D_ * D_);
    h16* w16_wq    = (h16*)alloc(sizeof(h16) * D_ * D_);
    h16* w16_wk    = (h16*)alloc(sizeof(h16) * D_ * D_);
    h16* w16_wv    = (h16*)alloc(sizeof(h16) * D_ * D_);
    h16* w16_wpos  = (h16*)alloc(sizeof(h16) * D_ * D_);
    h16* w16_wo    = (h16*)alloc(sizeof(h16) * D_ * D_);
    h16* w16_pw1   = (h16*)alloc(sizeof(h16) * D_ * 2 * D_);
    h16* w16_pw2   = (h16*)alloc(sizeof(h16) * D_ * D_);
    h16* w16_ff2w1 = (h16*)alloc(sizeof(h16) * D_ * 4 * D_);
    h16* w16_ff2w2 = (h16*)alloc(sizeof(h16) * 4 * D_ * D_);
    float* res   = (float*)alloc(sizeof(float) * MD);          // residual stream
    h16*   y16   = (h16*)alloc(sizeof(h16) * MD);              // LN output (f16)
    h16*   hid16 = (h16*)alloc(sizeof(h16) * MD * 4);          // FF hidden / pw1 out
    float* qf    = (float*)alloc(sizeof(float) * MD);          // q f32 (later glu)
    float* kf    = (float*)alloc(sizeof(float) * MD);          // k f32 (later conv16)
    float* vf    = (float*)alloc(sizeof(float) * MD);
    h16*   qu16  = (h16*)alloc(sizeof(h16) * MD);              // [B,H,S,DH]
    h16*   qv16  = (h16*)alloc(sizeof(h16) * MD);
    h16*   k16t  = (h16*)alloc(sizeof(h16) * MD);              // [B,H,S,DH]
    h16*   vT16  = (h16*)alloc(sizeof(h16) * MD);              // [B,H,DH,S]
    h16*   pe16  = (h16*)alloc(sizeof(h16) * S_ * D_);
    float* pfull = (float*)alloc(sizeof(float) * S_ * D_);
    h16*   p16   = (h16*)alloc(sizeof(h16) * S_ * D_);
    h16*   ps16  = (h16*)alloc(sizeof(h16) * (size_t)B_ * H_ * S_ * S_);
    h16*   ctx16 = (h16*)alloc(sizeof(h16) * MD);
    float* gluf  = qf;                                         // reuse
    h16*   conv16 = (h16*)kf;                                  // reuse

    auto blocks = [](long long n) { return (unsigned)((n + 255) / 256); };

    // ---- 0. convert weights to f16, transposed to [N][K] ----
    struct { const float* s; h16* d; int K; int N; } wc[] = {
        {ff1_w1, w16_ff1w1, D_, 4 * D_}, {ff1_w2, w16_ff1w2, 4 * D_, D_},
        {wq, w16_wq, D_, D_}, {wk, w16_wk, D_, D_}, {wv, w16_wv, D_, D_},
        {wpos, w16_wpos, D_, D_}, {wo, w16_wo, D_, D_},
        {pw1_w, w16_pw1, D_, 2 * D_}, {pw2_w, w16_pw2, D_, D_},
        {ff2_w1, w16_ff2w1, D_, 4 * D_}, {ff2_w2, w16_ff2w2, 4 * D_, D_}};
    for (auto& c : wc)
        f32_to_f16_T<<<blocks((long long)c.K * c.N), 256, 0, stream>>>(c.s, c.d, c.K, c.N);

    const long long SD  = (long long)S_ * DH_;      // 32768
    const long long HSD = (long long)H_ * SD;       // 262144
    const long long SS  = (long long)S_ * S_;       // 262144
    const long long HSS = (long long)H_ * SS;       // 2097152

    // ---- 1. Macaron FF1: res = x + 0.5*FF(LN(x)) ----
    ln_kernel<h16><<<MTOK, 256, 0, stream>>>(x, ff1_g, ff1_b, y16);
    gemm_wmma<h16, h16><<<dim3(4 * D_ / 64, MTOK / 128, 1), 256, 0, stream>>>(
        y16, w16_ff1w1, ff1_b1, nullptr, hid16, MTOK, 4 * D_, D_,
        D_, 0, 0, 0, 0, 4 * D_, 0, 0, 1, 0.f, 1);
    gemm_wmma<h16, float><<<dim3(D_ / 64, MTOK / 128, 1), 256, 0, stream>>>(
        hid16, w16_ff1w2, ff1_b2, x, res, MTOK, D_, 4 * D_,
        4 * D_, 0, 0, 0, 0, D_, 0, 0, 0, 0.5f, 1);

    // ---- 2. Rel-pos MHSA ----
    ln_kernel<h16><<<MTOK, 256, 0, stream>>>(res, attn_g, attn_b, y16);
    pe_kernel<<<blocks(S_ * D_), 256, 0, stream>>>(pe16, S_ * D_);
    gemm_wmma<h16, float><<<dim3(D_ / 64, MTOK / 128, 1), 256, 0, stream>>>(
        y16, w16_wq, bq, nullptr, qf, MTOK, D_, D_,
        D_, 0, 0, 0, 0, D_, 0, 0, 0, 0.f, 1);
    gemm_wmma<h16, float><<<dim3(D_ / 64, MTOK / 128, 1), 256, 0, stream>>>(
        y16, w16_wk, bk, nullptr, kf, MTOK, D_, D_,
        D_, 0, 0, 0, 0, D_, 0, 0, 0, 0.f, 1);
    gemm_wmma<h16, float><<<dim3(D_ / 64, MTOK / 128, 1), 256, 0, stream>>>(
        y16, w16_wv, bv, nullptr, vf, MTOK, D_, D_,
        D_, 0, 0, 0, 0, D_, 0, 0, 0, 0.f, 1);
    gemm_wmma<h16, float><<<dim3(D_ / 64, S_ / 128, 1), 256, 0, stream>>>(
        pe16, w16_wpos, nullptr, nullptr, pfull, S_, D_, D_,
        D_, 0, 0, 0, 0, D_, 0, 0, 0, 0.f, 1);
    qkv_prep<<<blocks(MD), 256, 0, stream>>>(qf, kf, vf, u_bias, v_bias,
                                             qu16, qv16, k16t, vT16, (int)MD);
    p_prep<<<blocks(H_ * S_ * DH_), 256, 0, stream>>>(pfull, p16, H_ * S_ * DH_);

    // content scores -> d_out attn region:  (q+u) @ K^T, per (b,h)
    gemm_wmma<h16, float><<<dim3(S_ / 64, S_ / 128, B_ * H_), 256, 0, stream>>>(
        qu16, k16t, nullptr, nullptr, attnOut, S_, S_, DH_,
        DH_, HSD, SD, HSD, SD, S_, HSS, SS, 0, 0.f, H_);
    // pos scores -> ps16 (f16):  (q+v) @ P^T, P shared over batch
    gemm_wmma<h16, h16><<<dim3(S_ / 64, S_ / 128, B_ * H_), 256, 0, stream>>>(
        qv16, p16, nullptr, nullptr, ps16, S_, S_, DH_,
        DH_, HSD, SD, 0, SD, S_, HSS, SS, 0, 0.f, H_);
    // softmax((content + rel_shift(pos)) / sqrt(D)) in place
    softmax_relshift<<<(unsigned)((long long)B_ * H_ * S_), 256, 0, stream>>>(
        attnOut, ps16, 1.0f / sqrtf((float)D_));
    // ctx = attn @ V  -> ctx16 [B,S,H,DH]   (B = vT16 [B,H,DH,S], row-major [n][k])
    gemm_wmma<float, h16><<<dim3(DH_ / 64, S_ / 128, B_ * H_), 256, 0, stream>>>(
        attnOut, vT16, nullptr, nullptr, ctx16, S_, DH_, S_,
        S_, HSS, SS, HSD, SD, D_, (long long)S_ * D_, DH_, 0, 0.f, H_);
    // out projection with residual: res += ctx @ wo + bo
    gemm_wmma<h16, float><<<dim3(D_ / 64, MTOK / 128, 1), 256, 0, stream>>>(
        ctx16, w16_wo, bo, res, res, MTOK, D_, D_,
        D_, 0, 0, 0, 0, D_, 0, 0, 0, 1.0f, 1);

    // ---- 3. Conv module ----
    ln_kernel<h16><<<MTOK, 256, 0, stream>>>(res, conv_g, conv_b, y16);
    gemm_wmma<h16, h16><<<dim3(2 * D_ / 64, MTOK / 128, 1), 256, 0, stream>>>(
        y16, w16_pw1, pw1_b, nullptr, hid16, MTOK, 2 * D_, D_,
        D_, 0, 0, 0, 0, 2 * D_, 0, 0, 0, 0.f, 1);
    glu_kernel<<<blocks(MD), 256, 0, stream>>>(hid16, gluf, (int)MD);
    dwconv_bn_swish<<<blocks(MD), 256, 0, stream>>>(gluf, dw_w, bn_g, bn_b, bn_m,
                                                    bn_v, conv16, (int)MD);
    gemm_wmma<h16, float><<<dim3(D_ / 64, MTOK / 128, 1), 256, 0, stream>>>(
        conv16, w16_pw2, pw2_b, res, res, MTOK, D_, D_,
        D_, 0, 0, 0, 0, D_, 0, 0, 0, 1.0f, 1);

    // ---- 4. Macaron FF2 + final LN ----
    ln_kernel<h16><<<MTOK, 256, 0, stream>>>(res, ff2_g, ff2_b, y16);
    gemm_wmma<h16, h16><<<dim3(4 * D_ / 64, MTOK / 128, 1), 256, 0, stream>>>(
        y16, w16_ff2w1, ff2_b1, nullptr, hid16, MTOK, 4 * D_, D_,
        D_, 0, 0, 0, 0, 4 * D_, 0, 0, 1, 0.f, 1);
    gemm_wmma<h16, float><<<dim3(D_ / 64, MTOK / 128, 1), 256, 0, stream>>>(
        hid16, w16_ff2w2, ff2_b2, res, res, MTOK, D_, 4 * D_,
        4 * D_, 0, 0, 0, 0, D_, 0, 0, 0, 0.5f, 1);
    ln_kernel<float><<<MTOK, 256, 0, stream>>>(res, ln_g, ln_b, outPtr);
}
